// KANAttention_38422777430222
// MI455X (gfx1250) — matrix operand order repaired and checked by reading, not compiled, not run on hardware
//
#include <hip/hip_runtime.h>
#include <hip/hip_bf16.h>

typedef _Float16 f16;
typedef __attribute__((ext_vector_type(16))) _Float16 v16h;
typedef __attribute__((ext_vector_type(8)))  float    v8f;
typedef __attribute__((ext_vector_type(4)))  int      v4i;

#define DIM 256
#define HEADS 8
#define HD 32
#define NTOK 2048
#define BATCH 2
#define ROWS (BATCH * NTOK)          // 4096
#define NFUNC 256

#if __has_builtin(__builtin_amdgcn_global_load_async_to_lds_b128) && \
    __has_builtin(__builtin_amdgcn_s_wait_asynccnt)
#define HAVE_ASYNC_LDS 1
#else
#define HAVE_ASYNC_LDS 0
#endif

// async-to-LDS builtin wants v4i pointers: global src in as(1), LDS dst in as(3)
#define GPTR4(p) ((__attribute__((address_space(1))) v4i*)(p))
#define LPTR4(p) ((__attribute__((address_space(3))) v4i*)(p))

// ---------- fragment load helpers (all 16B-aligned contiguous chunks) ----------
__device__ inline v16h ld_frag16(const f16* p) {      // 16 contiguous halves
    union { v16h v; int4 i[2]; } u;
    u.i[0] = *(const int4*)(p);
    u.i[1] = *(const int4*)(p + 8);
    return u.v;
}
__device__ inline v16h ld_frag8x2(const f16* p0, const f16* p1) { // two runs of 8
    union { v16h v; int4 i[2]; } u;
    u.i[0] = *(const int4*)(p0);
    u.i[1] = *(const int4*)(p1);
    return u.v;
}
__device__ inline v8f wmma_f16(v16h a, v16h b, v8f c) {
    return __builtin_amdgcn_wmma_f32_16x16x32_f16(false, a, false, b, (short)0, c,
                                                  false, false);
}

// ---------- prep kernels ----------
__global__ void cvt_f32_to_f16(const float* __restrict__ src, f16* __restrict__ dst, int n) {
    int i = blockIdx.x * blockDim.x + threadIdx.x;
    if (i < n) dst[i] = (f16)src[i];
}
// w[K][C] (row-major) -> wt[C][K] f16 (so B-fragments read contiguous K)
__global__ void transpose_w(const float* __restrict__ w, f16* __restrict__ wt, int K, int C) {
    int i = blockIdx.x * blockDim.x + threadIdx.x;
    if (i < K * C) {
        int k = i / C, c = i % C;
        wt[(size_t)c * K + k] = (f16)w[i];
    }
}

// ---------- QKV GEMM + fused KAN on K, scatter to per-head layouts ----------
// xh: [4096][256] f16, wT: [768][256] f16 (transposed w_qkv)
// Qh,Kh: [b][h][n][32] f16 ; Vt: [b][h][32][n] f16
__global__ __launch_bounds__(256) void qkv_kan_kernel(
    const f16* __restrict__ xh, const f16* __restrict__ wT,
    const float* __restrict__ ka, const float* __restrict__ kb,
    f16* __restrict__ Qh, f16* __restrict__ Kh, f16* __restrict__ Vt) {

    int wave = threadIdx.x >> 5, lane = threadIdx.x & 31;
    int tile = blockIdx.x * 8 + wave;            // 256 row-tiles * 48 col-tiles
    int rt = tile / 48, ct = tile % 48;
    int r0 = rt * 16, c0 = ct * 16;
    int ln = lane & 15, h = lane >> 4;

    v8f acc = {0.f,0.f,0.f,0.f,0.f,0.f,0.f,0.f};
    const f16* arow = xh + (size_t)(r0 + ln) * DIM;
    const f16* brow = wT + (size_t)(c0 + ln) * DIM;
#pragma unroll
    for (int kk = 0; kk < DIM; kk += 32) {
        v16h a = ld_frag16(arow + kk + 16 * h);
        v16h b = ld_frag16(brow + kk + 16 * h);
        acc = wmma_f16(a, b, acc);
    }

    // Fused KAN Fourier transform on the K section (cols 256..511):
    // sum_f a_f sin(f x)+b_f cos(f x) via angle-addition recurrence (4 FMA/step).
    if (c0 >= DIM && c0 < 2 * DIM) {
        float s[8], c_[8], s1[8], c1[8];
        v8f kan = {0.f,0.f,0.f,0.f,0.f,0.f,0.f,0.f};
#pragma unroll
        for (int g = 0; g < 8; ++g) {
            float xv = acc[g];
            s1[g] = __sinf(xv); c1[g] = __cosf(xv);
            s[g] = s1[g]; c_[g] = c1[g];
        }
        for (int f = 0; f < NFUNC; ++f) {
            float af = ka[f], bf = kb[f];
#pragma unroll
            for (int g = 0; g < 8; ++g) {
                kan[g] = fmaf(af, s[g], fmaf(bf, c_[g], kan[g]));
                float ns = fmaf(s[g], c1[g],  c_[g] * s1[g]);
                float nc = fmaf(c_[g], c1[g], -s[g] * s1[g]);
                s[g] = ns; c_[g] = nc;
            }
        }
        acc = kan;
    }

    // scatter into per-head layouts
#pragma unroll
    for (int g = 0; g < 8; ++g) {
        int r = r0 + g + 8 * h;                  // global row 0..4095
        int c = c0 + ln;                         // qkv col 0..767
        int b = r >> 11, n = r & (NTOK - 1);
        f16 val = (f16)acc[g];
        if (c < DIM) {
            int head = c >> 5, hd = c & 31;
            Qh[(((size_t)b * HEADS + head) * NTOK + n) * HD + hd] = val;
        } else if (c < 2 * DIM) {
            int cp = c - DIM, head = cp >> 5, hd = cp & 31;
            Kh[(((size_t)b * HEADS + head) * NTOK + n) * HD + hd] = val;
        } else {
            int cp = c - 2 * DIM, head = cp >> 5, hd = cp & 31;
            Vt[(((size_t)b * HEADS + head) * HD + hd) * NTOK + n] = val;
        }
    }
}

// ---------- flash attention ----------
// 4 waves/block, all on the SAME (b,h): K/V chunks are staged once per block
// into LDS with double-buffered async global->LDS copies, then all 4 waves
// read their WMMA fragments from LDS.
// Qh,Kh: [bh][n][32] ; Vt: [bh][32][n] ; Oc: [b][n][256] f16
__global__ __launch_bounds__(128) void flash_attn_kernel(
    const f16* __restrict__ Qh, const f16* __restrict__ Kh,
    const f16* __restrict__ Vt, f16* __restrict__ Oc) {

    __shared__ __align__(16) f16 sK[2][32 * HD];   // 32 keys x 32 hd   (2 KB/buf)
    __shared__ __align__(16) f16 sV[2][HD * 32];   // 32 hd  x 32 keys  (2 KB/buf)
    __shared__ __align__(16) f16 sP[4][16 * 32];   // per-wave P staging tile

    int wave = threadIdx.x >> 5, lane = threadIdx.x & 31;
    int bh = blockIdx.x >> 5;                     // 16 (b,h) pairs, 32 blocks each
    int q0 = ((blockIdx.x & 31) * 4 + wave) * 16; // this wave's 16-query tile
    int ln = lane & 15, h = lane >> 4;
    int tid = threadIdx.x;                        // 0..127

    const f16* Qb = Qh + (size_t)bh * NTOK * HD;
    const f16* Kb = Kh + (size_t)bh * NTOK * HD;
    const f16* Vb = Vt + (size_t)bh * HD * NTOK;

    const f16* qrow = Qb + (size_t)(q0 + ln) * HD;
    v16h qa = ld_frag16(qrow + 16 * h);

    v8f o0 = {0.f,0.f,0.f,0.f,0.f,0.f,0.f,0.f};
    v8f o1 = {0.f,0.f,0.f,0.f,0.f,0.f,0.f,0.f};
    float m[8], l[8];
#pragma unroll
    for (int g = 0; g < 8; ++g) { m[g] = -1e30f; l[g] = 0.f; }
    const float scale = 0.17677669529663687f;     // 1/sqrt(32)
    f16* sp = sP[wave];

    // cooperative 32-key chunk stage: 128 threads x 16B = 2KB for K, 2KB for V
    int vrow = tid >> 2, vseg = tid & 3;          // V: 32 rows x 4 segs of 8 halves
    auto stage = [&](int buf, int kc) {
        const f16* gk = Kb + (size_t)kc * HD + tid * 8;              // contiguous
        const f16* gv = Vb + (size_t)vrow * NTOK + kc + vseg * 8;    // per-hd row
        f16* dk = sK[buf] + tid * 8;
        f16* dv = sV[buf] + vrow * 32 + vseg * 8;
#if HAVE_ASYNC_LDS
        __builtin_amdgcn_global_load_async_to_lds_b128(GPTR4(gk), LPTR4(dk), 0, 0);
        __builtin_amdgcn_global_load_async_to_lds_b128(GPTR4(gv), LPTR4(dv), 0, 0);
#else
        *(int4*)dk = *(const int4*)gk;
        *(int4*)dv = *(const int4*)gv;
#endif
    };

    stage(0, 0);
    for (int kc = 0; kc < NTOK; kc += 32) {
        int buf = (kc >> 5) & 1;
        bool pf = (kc + 32 < NTOK);
        if (pf) stage(buf ^ 1, kc + 32);          // prefetch next chunk
#if HAVE_ASYNC_LDS
        if (pf) __builtin_amdgcn_s_wait_asynccnt(2);   // older pair (this buf) done
        else    __builtin_amdgcn_s_wait_asynccnt(0);
#endif
        __syncthreads();                          // all waves staged this buf

        // S = Q * K^T for 32 keys (two 16-col B fragments from LDS)
        const f16* sk = sK[buf];
        const f16* sv = sV[buf];
        v16h kb0 = ld_frag16(sk + (size_t)ln * HD + 16 * h);
        v16h kb1 = ld_frag16(sk + (size_t)(16 + ln) * HD + 16 * h);
        v8f z = {0.f,0.f,0.f,0.f,0.f,0.f,0.f,0.f};
        v8f s0 = wmma_f16(qa, kb0, z);
        v8f s1 = wmma_f16(qa, kb1, z);

        // online softmax per row (row = g + 8h, stats reduced over 16-lane half)
#pragma unroll
        for (int g = 0; g < 8; ++g) {
            float a = s0[g] * scale, b2 = s1[g] * scale;
            float rmax = fmaxf(a, b2);
#pragma unroll
            for (int mk = 1; mk < 16; mk <<= 1)
                rmax = fmaxf(rmax, __shfl_xor(rmax, mk, 16));
            float mn = fmaxf(m[g], rmax);
            float corr = __expf(m[g] - mn);
            float p0 = __expf(a - mn), p1 = __expf(b2 - mn);
            float rs = p0 + p1;
#pragma unroll
            for (int mk = 1; mk < 16; mk <<= 1)
                rs += __shfl_xor(rs, mk, 16);
            l[g] = l[g] * corr + rs;
            m[g] = mn;
            o0[g] *= corr; o1[g] *= corr;
            int row = g + 8 * h;
            sp[row * 32 + ln]      = (f16)p0;     // LDS bounce: C/D -> A layout
            sp[row * 32 + 16 + ln] = (f16)p1;
        }
        // P as A-fragment (16x32), V as two B-fragments (keys contiguous in sV)
        const f16* prow = sp + ln * 32;
        v16h pa  = ld_frag16(prow + 16 * h);
        v16h vb0 = ld_frag16(sv + (size_t)ln * 32 + 16 * h);
        v16h vb1 = ld_frag16(sv + (size_t)(16 + ln) * 32 + 16 * h);
        o0 = wmma_f16(pa, vb0, o0);
        o1 = wmma_f16(pa, vb1, o1);

        __syncthreads();                          // all done reading before overwrite
    }

    int b = bh >> 3, d0 = (bh & 7) * HD;
#pragma unroll
    for (int g = 0; g < 8; ++g) {
        float inv = 1.0f / l[g];
        int n = q0 + g + 8 * h;
        size_t base = ((size_t)b * NTOK + n) * DIM + d0;
        Oc[base + ln]      = (f16)(o0[g] * inv);
        Oc[base + 16 + ln] = (f16)(o1[g] * inv);
    }
}

// ---------- output projection: Oc[4096][256] @ w_out + b_out -> f32 ----------
__global__ __launch_bounds__(256) void outproj_kernel(
    const f16* __restrict__ Oc, const f16* __restrict__ wT2,
    const float* __restrict__ bias, float* __restrict__ out) {

    int wave = threadIdx.x >> 5, lane = threadIdx.x & 31;
    int tile = blockIdx.x * 8 + wave;             // 256 row-tiles * 16 col-tiles
    int rt = tile >> 4, ct = tile & 15;
    int r0 = rt * 16, c0 = ct * 16;
    int ln = lane & 15, h = lane >> 4;

    v8f acc = {0.f,0.f,0.f,0.f,0.f,0.f,0.f,0.f};
    const f16* arow = Oc + (size_t)(r0 + ln) * DIM;
    const f16* brow = wT2 + (size_t)(c0 + ln) * DIM;
#pragma unroll
    for (int kk = 0; kk < DIM; kk += 32) {
        v16h a = ld_frag16(arow + kk + 16 * h);
        v16h b = ld_frag16(brow + kk + 16 * h);
        acc = wmma_f16(a, b, acc);
    }
    float bv = bias[c0 + ln];
#pragma unroll
    for (int g = 0; g < 8; ++g) {
        int r = r0 + g + 8 * h;
        out[(size_t)r * DIM + c0 + ln] = acc[g] + bv;
    }
}

extern "C" void kernel_launch(void* const* d_in, const int* in_sizes, int n_in,
                              void* d_out, int out_size, void* d_ws, size_t ws_size,
                              hipStream_t stream) {
    const float* x     = (const float*)d_in[0];   // [2,2048,256]
    const float* w_qkv = (const float*)d_in[1];   // [256,768]
    const float* w_out = (const float*)d_in[2];   // [256,256]
    const float* b_out = (const float*)d_in[3];   // [256]
    const float* kan_a = (const float*)d_in[4];   // [256]
    const float* kan_b = (const float*)d_in[5];   // [256]
    float* out = (float*)d_out;                   // [2,2048,256] f32

    char* ws = (char*)d_ws;
    size_t off = 0;
    f16* xh  = (f16*)(ws + off); off += (size_t)ROWS * DIM * 2;        // 2 MB
    f16* wT  = (f16*)(ws + off); off += (size_t)DIM * 3 * DIM * 2;     // 384 KB
    f16* wT2 = (f16*)(ws + off); off += (size_t)DIM * DIM * 2;         // 128 KB
    f16* Qh  = (f16*)(ws + off); off += (size_t)ROWS * DIM * 2;        // 2 MB
    f16* Kh  = (f16*)(ws + off); off += (size_t)ROWS * DIM * 2;        // 2 MB
    f16* Vt  = (f16*)(ws + off); off += (size_t)ROWS * DIM * 2;        // 2 MB
    f16* Oc  = (f16*)(ws + off); off += (size_t)ROWS * DIM * 2;        // 2 MB
    (void)ws_size; (void)in_sizes; (void)n_in; (void)out_size;

    // prep: f32->f16 conversions + weight transposes
    {
        int n = ROWS * DIM;
        cvt_f32_to_f16<<<(n + 255) / 256, 256, 0, stream>>>(x, xh, n);
        transpose_w<<<(DIM * 3 * DIM + 255) / 256, 256, 0, stream>>>(w_qkv, wT, DIM, 3 * DIM);
        transpose_w<<<(DIM * DIM + 255) / 256, 256, 0, stream>>>(w_out, wT2, DIM, DIM);
    }
    // QKV GEMM + fused KAN (12288 tiles, 8 waves/block)
    qkv_kan_kernel<<<12288 / 8, 256, 0, stream>>>(xh, wT, kan_a, kan_b, Qh, Kh, Vt);
    // flash attention (512 blocks x 4 waves, block shares one (b,h))
    flash_attn_kernel<<<512, 128, 0, stream>>>(Qh, Kh, Vt, Oc);
    // output projection (4096 tiles, 8 waves/block)
    outproj_kernel<<<4096 / 8, 256, 0, stream>>>(Oc, wT2, b_out, out);
}
